// MultiHeadAttention_52587579572330
// MI455X (gfx1250) — matrix operand order repaired and checked by reading, not compiled, not run on hardware
//
#include <hip/hip_runtime.h>

// ---------------- problem constants (from reference) ----------------
#define H_    16
#define D_    128
#define B_    2
#define NSEQ  2048
#define CDIM  2048
#define BN    (B_ * NSEQ)   // 4096 rows
#define N3    (3 * CDIM)    // 6144 qkv cols
#define QSCALE 0.08838834764831845f   // 1/sqrt(128)
#define EPS_   1.1920929e-7f

typedef __attribute__((ext_vector_type(16))) __bf16        v16bf;
typedef __attribute__((ext_vector_type(8)))  float         v8f;
typedef __attribute__((ext_vector_type(4)))  unsigned int  u32x4;

union ABfrag { v16bf v; u32x4 u[2]; };

__device__ __forceinline__ v8f v8f_zero() {
  v8f z;
#pragma unroll
  for (int i = 0; i < 8; ++i) z[i] = 0.0f;
  return z;
}

// ds_swizzle xor-shuffle (group-of-32 encoding: and=0x1f, or=0, xor=XM)
#define SWZ_XOR(v, XM) __builtin_bit_cast(float, \
    __builtin_amdgcn_ds_swizzle(__builtin_bit_cast(int, (v)), (((XM) << 10) | 0x1f)))

// reductions across each 16-lane half (xor 1,2,4,8 stays inside the half)
__device__ __forceinline__ float rmax16(float v) {
  v = fmaxf(v, SWZ_XOR(v, 1));
  v = fmaxf(v, SWZ_XOR(v, 2));
  v = fmaxf(v, SWZ_XOR(v, 4));
  v = fmaxf(v, SWZ_XOR(v, 8));
  return v;
}
__device__ __forceinline__ float rsum16(float v) {
  v += SWZ_XOR(v, 1);
  v += SWZ_XOR(v, 2);
  v += SWZ_XOR(v, 4);
  v += SWZ_XOR(v, 8);
  return v;
}
__device__ __forceinline__ float rsum32(float v) {
  v += SWZ_XOR(v, 1);
  v += SWZ_XOR(v, 2);
  v += SWZ_XOR(v, 4);
  v += SWZ_XOR(v, 8);
  v += SWZ_XOR(v, 16);
  return v;
}

// ---------------- fp32 -> bf16 conversion ----------------
__global__ __launch_bounds__(256) void cvt_bf16_kern(const float* __restrict__ src,
                                                     __bf16* __restrict__ dst, int n) {
  int i = blockIdx.x * blockDim.x + threadIdx.x;
  int stride = gridDim.x * blockDim.x;
  for (; i < n; i += stride) dst[i] = (__bf16)src[i];
}

// ---------------- TN GEMM: C[M,N] = A[M,K] * B[N,K]^T + bias ----------------
// block = 256 threads = 8 waves; block tile 64 rows x 256 cols;
// each wave: 16 rows x 128 cols (8 WMMA accumulators).
// K is a compile-time constant so all 8 B-tile addresses are immediate
// offsets off a single pointer; two-deep register double buffer over K
// (step 32, unroll 2, last iteration peeled -> no wrap select, no branches).
// mode 0: store f32 to outF [M, CDIM]
// mode 1: qkv epilogue -> scatter bf16 into q[bh][n][d], k[bh][n][d], vt[bh][d][n]
template <int K>
__global__ __launch_bounds__(256) void gemm_tn_kern(
    const __bf16* __restrict__ A, const __bf16* __restrict__ Bm,
    const float* __restrict__ bias, int mode,
    float* __restrict__ outF,
    __bf16* __restrict__ qbuf, __bf16* __restrict__ kbuf, __bf16* __restrict__ vtbuf) {
  const int lane    = threadIdx.x & 31;
  const int wave    = threadIdx.x >> 5;
  const int l16     = lane & 15;
  const int half    = lane >> 4;                 // 0: lanes 0-15, 1: lanes 16-31
  const int rowBase = blockIdx.x * 64 + (wave & 3) * 16;
  const int colBase = blockIdx.y * 256 + (wave >> 2) * 128;

  v8f acc[8];
#pragma unroll
  for (int t = 0; t < 8; ++t) acc[t] = v8f_zero();

  // A operand: row = rowBase + l16; lane<16 gets K 0..15, lane>=16 K 16..31
  const __bf16* aP = A + (size_t)(rowBase + l16) * K + half * 16;
  // B operand: col = colBase + t*16 + l16; lane<16 K 0..15, lane>=16 K 16..31
  // tile t lives at constant element offset t*16*K from bP (immediate-foldable)
  const __bf16* bP = Bm + (size_t)(colBase + l16) * K + half * 16;

  ABfrag a0, a1, b0[8], b1[8];

  auto loadk = [&](ABfrag& aF, ABfrag (&bF)[8], int koff) {
    aF.u[0] = *(const u32x4*)(aP + koff);
    aF.u[1] = *(const u32x4*)(aP + koff + 8);
#pragma unroll
    for (int t = 0; t < 8; ++t) {
      bF[t].u[0] = *(const u32x4*)(bP + t * 16 * K + koff);
      bF[t].u[1] = *(const u32x4*)(bP + t * 16 * K + koff + 8);
    }
  };
  auto compute = [&](const ABfrag& aF, const ABfrag (&bF)[8]) {
#pragma unroll
    for (int t = 0; t < 8; ++t)
      acc[t] = __builtin_amdgcn_wmma_f32_16x16x32_bf16(
          false, aF.v, false, bF[t].v, (short)0, acc[t], false, false);
  };

  loadk(a0, b0, 0);
#pragma unroll 1
  for (int it = 0; it < K / 64 - 1; ++it) {
    loadk(a1, b1, 32);      // odd chunk of current 64-block
    compute(a0, b0);
    aP += 64;
    bP += 64;
    loadk(a0, b0, 0);       // even chunk of next 64-block
    compute(a1, b1);
  }
  loadk(a1, b1, 32);        // peeled tail
  compute(a0, b0);
  compute(a1, b1);

#pragma unroll
  for (int t = 0; t < 8; ++t) {
    const int col = colBase + t * 16 + l16;
    const float bv = bias[col];
#pragma unroll
    for (int r = 0; r < 8; ++r) {
      const int row = rowBase + r + half * 8;     // C layout: VGPR r -> M=r / r+8
      const float v = acc[t][r] + bv;
      if (mode == 0) {
        outF[(size_t)row * CDIM + col] = v;
      } else {
        const int which = col >> 11;              // 0=q 1=k 2=v (uniform per wave tile)
        const int hd = col & 2047;
        const int h = hd >> 7, d = hd & 127;
        const int bidx = row >> 11, nq = row & 2047;
        const int bh = bidx * H_ + h;
        const __bf16 bvv = (__bf16)v;
        if (which == 0)      qbuf[((size_t)bh * NSEQ + nq) * D_ + d] = bvv;
        else if (which == 1) kbuf[((size_t)bh * NSEQ + nq) * D_ + d] = bvv;
        else                 vtbuf[((size_t)bh * D_ + d) * NSEQ + nq] = bvv;  // transposed
      }
    }
  }
}

// ---------------- RMSNorm + RoPE (in place on bf16 [bh][n][128]) ----------------
// one wave per row; lane owns d = {lane, lane+32, lane+64, lane+96}
__global__ __launch_bounds__(256) void norm_rope_kern(
    __bf16* __restrict__ t, const float* __restrict__ rope,
    const float* __restrict__ g, float outScale) {
  const int lane = threadIdx.x & 31;
  const int wave = threadIdx.x >> 5;
  const int row  = blockIdx.x * 8 + wave;         // bh*2048 + n
  const int n    = row & (NSEQ - 1);
  __bf16* p = t + (size_t)row * D_;

  float t0 = (float)p[lane];
  float t1 = (float)p[lane + 32];
  float t2 = (float)p[lane + 64];
  float t3 = (float)p[lane + 96];

  float ss = rsum32(t0 * t0 + t1 * t1 + t2 * t2 + t3 * t3);
  float inv = rsqrtf(ss * (1.0f / 128.0f) + EPS_);

  float a1 = t0 * inv * g[lane],      a2 = t2 * inv * g[lane + 64];
  float b1 = t1 * inv * g[lane + 32], b2 = t3 * inv * g[lane + 96];

  float ang0 = rope[(size_t)n * 64 + lane];
  float ang1 = rope[(size_t)n * 64 + lane + 32];
  float c0 = __cosf(ang0), s0 = __sinf(ang0);
  float c1 = __cosf(ang1), s1 = __sinf(ang1);

  p[lane]      = (__bf16)((a1 * c0 - a2 * s0) * outScale);
  p[lane + 64] = (__bf16)((a2 * c0 + a1 * s0) * outScale);
  p[lane + 32] = (__bf16)((b1 * c1 - b2 * s1) * outScale);
  p[lane + 96] = (__bf16)((b2 * c1 + b1 * s1) * outScale);
}

// ---------------- flash attention ----------------
// grid: x = N/64 query blocks, y = B*H; block = 128 threads = 4 waves;
// each wave: 16 queries x D=128, keys in tiles of 32.
// Two K-fragment buffers give cross-iteration prefetch: the next tile's K
// loads issue right after the current S-WMMAs, hidden behind softmax VALU +
// LDS transpose + PV-WMMAs. V loads issue at tile start, hidden behind the
// S-WMMAs and softmax. Last tile peeled -> no conditional prefetch.
__global__ __launch_bounds__(128) void attn_kern(
    const __bf16* __restrict__ q, const __bf16* __restrict__ k,
    const __bf16* __restrict__ vt, __bf16* __restrict__ out) {
  __shared__ __align__(16) __bf16 plds[4 * 16 * 32];   // per-wave 16x32 P tile

  const int lane  = threadIdx.x & 31;
  const int wave  = threadIdx.x >> 5;
  const int l16   = lane & 15;
  const int half  = lane >> 4;
  const int bh    = blockIdx.y;
  const int bidx  = bh >> 4, h = bh & 15;
  const int qBase = blockIdx.x * 64 + wave * 16;

  // resident Q fragments (A operand), 4 chunks of 32 over D=128
  ABfrag qf[4];
  {
    const __bf16* qrow = q + ((size_t)bh * NSEQ + qBase + l16) * D_ + half * 16;
#pragma unroll
    for (int c = 0; c < 4; ++c) {
      qf[c].u[0] = *(const u32x4*)(qrow + c * 32);
      qf[c].u[1] = *(const u32x4*)(qrow + c * 32 + 8);
    }
  }

  float m[8], l[8], al[8];
#pragma unroll
  for (int r = 0; r < 8; ++r) { m[r] = -1e30f; l[r] = 0.0f; }
  v8f acc[8];
#pragma unroll
  for (int t = 0; t < 8; ++t) acc[t] = v8f_zero();

  __bf16* pw = plds + wave * 512;
  const __bf16* krowBase = k + (size_t)bh * NSEQ * D_ + (size_t)l16 * D_ + half * 16;
  const __bf16* vrowBase = vt + (size_t)bh * D_ * NSEQ + (size_t)l16 * NSEQ + half * 16;

  auto loadK = [&](ABfrag (&kf)[8], int kt) {
    const __bf16* kc = krowBase + (size_t)kt * D_;
#pragma unroll
    for (int c = 0; c < 4; ++c) {
      kf[2 * c].u[0]     = *(const u32x4*)(kc + c * 32);
      kf[2 * c].u[1]     = *(const u32x4*)(kc + c * 32 + 8);
      kf[2 * c + 1].u[0] = *(const u32x4*)(kc + 16 * D_ + c * 32);
      kf[2 * c + 1].u[1] = *(const u32x4*)(kc + 16 * D_ + c * 32 + 8);
    }
  };

  // one 32-key tile: S-WMMAs with kcur, optionally prefetch knext, softmax,
  // LDS C->A transpose, PV-WMMAs
  auto step = [&](ABfrag (&kcur)[8], ABfrag (&knext)[8], int ktThis, int ktNext,
                  bool doPrefetch) {
    // V fragments for this tile (consumed only after softmax)
    ABfrag vf[8];
#pragma unroll
    for (int t = 0; t < 8; ++t) {
      const __bf16* vc = vrowBase + (size_t)(t * 16) * NSEQ + ktThis;
      vf[t].u[0] = *(const u32x4*)(vc);
      vf[t].u[1] = *(const u32x4*)(vc + 8);
    }

    v8f s0 = v8f_zero(), s1 = v8f_zero();
#pragma unroll
    for (int c = 0; c < 4; ++c) {
      s0 = __builtin_amdgcn_wmma_f32_16x16x32_bf16(false, qf[c].v, false, kcur[2 * c].v,
                                                   (short)0, s0, false, false);
      s1 = __builtin_amdgcn_wmma_f32_16x16x32_bf16(false, qf[c].v, false, kcur[2 * c + 1].v,
                                                   (short)0, s1, false, false);
    }

    if (doPrefetch) loadK(knext, ktNext);   // hidden behind softmax + LDS + PV

    // online softmax: row r lives in VGPR r across each 16-lane half
#pragma unroll
    for (int r = 0; r < 8; ++r) {
      float mx = rmax16(fmaxf(s0[r], s1[r]));
      float mn = fmaxf(m[r], mx);
      al[r] = __expf(m[r] - mn);
      float p0 = __expf(s0[r] - mn);
      float p1 = __expf(s1[r] - mn);
      l[r] = l[r] * al[r] + rsum16(p0 + p1);
      m[r] = mn;
      s0[r] = p0; s1[r] = p1;
    }
#pragma unroll
    for (int t = 0; t < 8; ++t)
#pragma unroll
      for (int r = 0; r < 8; ++r) acc[t][r] *= al[r];

    // C-layout -> A-operand layout via per-wave LDS tile
#pragma unroll
    for (int r = 0; r < 8; ++r) {
      const int prow = r + half * 8;
      pw[prow * 32 + l16]      = (__bf16)s0[r];
      pw[prow * 32 + 16 + l16] = (__bf16)s1[r];
    }
    __syncthreads();
    ABfrag pa;
    {
      const __bf16* pr = pw + l16 * 32 + half * 16;
      pa.u[0] = *(const u32x4*)(pr);
      pa.u[1] = *(const u32x4*)(pr + 8);
    }
    __syncthreads();

#pragma unroll
    for (int t = 0; t < 8; ++t)
      acc[t] = __builtin_amdgcn_wmma_f32_16x16x32_bf16(false, pa.v, false, vf[t].v,
                                                       (short)0, acc[t], false, false);
  };

  ABfrag kfA[8], kfB[8];
  loadK(kfA, 0);
#pragma unroll 1
  for (int kt = 0; kt < NSEQ - 64; kt += 64) {
    step(kfA, kfB, kt, kt + 32, true);
    step(kfB, kfA, kt + 32, kt + 64, true);
  }
  step(kfA, kfB, NSEQ - 64, NSEQ - 32, true);   // peeled tail
  step(kfB, kfA, NSEQ - 32, 0, false);

  // epilogue: O /= l, scatter to attn_out[b][n][h*128 + d] as bf16
#pragma unroll
  for (int r = 0; r < 8; ++r) {
    const float invl = 1.0f / l[r];
    const int nq = qBase + r + half * 8;
    __bf16* orow = out + ((size_t)(bidx * NSEQ + nq)) * CDIM + h * D_;
#pragma unroll
    for (int t = 0; t < 8; ++t)
      orow[t * 16 + l16] = (__bf16)(acc[t][r] * invl);
  }
}

// ---------------- launcher ----------------
extern "C" void kernel_launch(void* const* d_in, const int* in_sizes, int n_in,
                              void* d_out, int out_size, void* d_ws, size_t ws_size,
                              hipStream_t stream) {
  const float* x        = (const float*)d_in[0];
  const float* rope     = (const float*)d_in[1];
  const float* qkv_w    = (const float*)d_in[2];
  const float* qkv_b    = (const float*)d_in[3];
  const float* proj_w   = (const float*)d_in[4];
  const float* proj_b   = (const float*)d_in[5];
  const float* q_norm_w = (const float*)d_in[6];
  const float* k_norm_w = (const float*)d_in[7];

  char* ws = (char*)d_ws;
  size_t off = 0;
  auto alloc = [&](size_t bytes) -> void* {
    void* p = ws + off;
    off += (bytes + 255) & ~(size_t)255;
    return p;
  };
  __bf16* xb    = (__bf16*)alloc((size_t)BN * CDIM * 2);            // 16 MB
  __bf16* wqkv  = (__bf16*)alloc((size_t)N3 * CDIM * 2);            // 24 MB
  __bf16* wproj = (__bf16*)alloc((size_t)CDIM * CDIM * 2);          //  8 MB
  __bf16* qb    = (__bf16*)alloc((size_t)B_ * H_ * NSEQ * D_ * 2);  // 16 MB
  __bf16* kb    = (__bf16*)alloc((size_t)B_ * H_ * NSEQ * D_ * 2);  // 16 MB
  __bf16* vtb   = (__bf16*)alloc((size_t)B_ * H_ * NSEQ * D_ * 2);  // 16 MB
  __bf16* attnb = (__bf16*)alloc((size_t)BN * CDIM * 2);            // 16 MB

  cvt_bf16_kern<<<4096, 256, 0, stream>>>(x, xb, BN * CDIM);
  cvt_bf16_kern<<<4096, 256, 0, stream>>>(qkv_w, wqkv, N3 * CDIM);
  cvt_bf16_kern<<<4096, 256, 0, stream>>>(proj_w, wproj, CDIM * CDIM);

  dim3 g1(BN / 64, N3 / 256);
  gemm_tn_kern<CDIM><<<g1, 256, 0, stream>>>(xb, wqkv, qkv_b, 1,
                                             nullptr, qb, kb, vtb);

  norm_rope_kern<<<(B_ * H_ * NSEQ) / 8, 256, 0, stream>>>(qb, rope, q_norm_w, QSCALE);
  norm_rope_kern<<<(B_ * H_ * NSEQ) / 8, 256, 0, stream>>>(kb, rope, k_norm_w, 1.0f);

  dim3 g2(NSEQ / 64, B_ * H_);
  attn_kern<<<g2, 128, 0, stream>>>(qb, kb, vtb, attnb);

  dim3 g3(BN / 64, CDIM / 256);
  gemm_tn_kern<CDIM><<<g3, 256, 0, stream>>>(attnb, wproj, proj_b, 0,
                                             (float*)d_out, nullptr, nullptr, nullptr);
}
